// LSTMModel_18408229831246
// MI455X (gfx1250) — matrix operand order repaired and checked by reading, not compiled, not run on hardware
//
#include <hip/hip_runtime.h>

// ---------------- problem constants ----------------
#define S_LEN  512
#define BATCH  64
#define DIN0   132
#define DIN0P  160      // DIN0 padded up to a multiple of 32 (WMMA K granularity)
#define HID    1024
#define G4     4096     // 4*HID gate width
#define NTILES 256      // G4 / 16 column tiles
#define NWG    64       // persistent workgroups (all co-resident; trivially fits)
#define NTHR   256      // 8 waves of 32 (wave32)
#define EPS_LN 1e-5f

typedef __attribute__((ext_vector_type(16))) __bf16 bf16x16;
typedef __attribute__((ext_vector_type(8)))  float  f32x8;
typedef __attribute__((ext_vector_type(8)))  unsigned short u16x8;
typedef __attribute__((ext_vector_type(4)))  float  f32x4;

union AB32 { bf16x16 v; u16x8 h[2]; };

#define WMMA_BF16(A, B, C) \
  __builtin_amdgcn_wmma_f32_16x16x32_bf16(false, (A), false, (B), (short)0, (C), false, false)

// ---- A-operand (16x32 bf16) per-lane fragment, per CDNA5 ISA layout:
// lane<16: row=lane, K = {0..7}U{16..23}; lane>=16: row=lane-16, K shifted +8.
__device__ inline bf16x16 load_a_bf16(const __bf16* __restrict__ base, int stride,
                                      int kt, int lane) {
  const int row = lane & 15, hs = lane >> 4;
  const __bf16* p = base + (size_t)row * stride + kt * 32 + hs * 8;
  AB32 a;
  a.h[0] = *reinterpret_cast<const u16x8*>(p);        // elems 0..7  -> K base+0..7
  a.h[1] = *reinterpret_cast<const u16x8*>(p + 16);   // elems 8..15 -> K base+16..23
  return a.v;
}

__device__ inline bf16x16 load_a_f32(const float* __restrict__ base, int stride,
                                     int kt, int lane) {
  const int row = lane & 15, hs = lane >> 4;
  const float* p = base + (size_t)row * stride + kt * 32 + hs * 8;
  f32x4 f0 = *reinterpret_cast<const f32x4*>(p);
  f32x4 f1 = *reinterpret_cast<const f32x4*>(p + 4);
  f32x4 f2 = *reinterpret_cast<const f32x4*>(p + 16);
  f32x4 f3 = *reinterpret_cast<const f32x4*>(p + 20);
  bf16x16 a;
#pragma unroll
  for (int e = 0; e < 4; ++e) {
    a[e]      = (__bf16)f0[e];
    a[e + 4]  = (__bf16)f1[e];
    a[e + 8]  = (__bf16)f2[e];
    a[e + 12] = (__bf16)f3[e];
  }
  return a;
}

// ---- B-operand: weights pre-tiled so each lane's 16 bf16 are contiguous (32B).
__device__ inline bf16x16 load_b(const __bf16* __restrict__ wt, int kt, int nt, int lane) {
  const __bf16* p = wt + ((((size_t)kt * NTILES + nt) * 32 + lane) << 4);
  AB32 b;
  b.h[0] = *reinterpret_cast<const u16x8*>(p);
  b.h[1] = *reinterpret_cast<const u16x8*>(p + 8);
  return b.v;
}

// ---- C/D layout: VGPR r, lanes0-15 -> M=r, lanes16-31 -> M=r+8; N = lane&15.
__device__ inline void store_c(const f32x8& acc, float* __restrict__ raw,
                               int m, int nt, int lane) {
  const int hs = lane >> 4, col = nt * 16 + (lane & 15);
#pragma unroll
  for (int r = 0; r < 8; ++r) {
    int brow = m * 16 + r + 8 * hs;
    raw[(size_t)brow * G4 + col] = acc[r];
  }
}

// One wave: two 16x16 output tiles over K = kts*32, raw (pre-LN) result to `raw`.
__device__ inline void gemm2(const __bf16* __restrict__ wt, float* __restrict__ raw,
                             const __bf16* __restrict__ abf, const float* __restrict__ af32,
                             int astride, int kts, int m, int nt0, int lane) {
  f32x8 acc0 = {0, 0, 0, 0, 0, 0, 0, 0};
  f32x8 acc1 = {0, 0, 0, 0, 0, 0, 0, 0};
  if (abf) {
    for (int kt = 0; kt < kts; ++kt) {
      bf16x16 a  = load_a_bf16(abf, astride, kt, lane);
      bf16x16 b0 = load_b(wt, kt, nt0,     lane);
      bf16x16 b1 = load_b(wt, kt, nt0 + 1, lane);
      acc0 = WMMA_BF16(a, b0, acc0);
      acc1 = WMMA_BF16(a, b1, acc1);
    }
  } else {
    for (int kt = 0; kt < kts; ++kt) {
      bf16x16 a  = load_a_f32(af32, astride, kt, lane);
      bf16x16 b0 = load_b(wt, kt, nt0,     lane);
      bf16x16 b1 = load_b(wt, kt, nt0 + 1, lane);
      acc0 = WMMA_BF16(a, b0, acc0);
      acc1 = WMMA_BF16(a, b1, acc1);
    }
  }
  store_c(acc0, raw, m, nt0,     lane);
  store_c(acc1, raw, m, nt0 + 1, lane);
}

// ---- device-wide monotonic barrier (64 resident WGs) ----
__device__ inline void grid_sync(unsigned* bar, unsigned target) {
  __syncthreads();
  if (threadIdx.x == 0) {
    __threadfence();                       // publish this WG's writes
    atomicAdd(bar, 1u);
    while (__hip_atomic_load(bar, __ATOMIC_ACQUIRE, __HIP_MEMORY_SCOPE_AGENT) < target)
      __builtin_amdgcn_s_sleep(2);
  }
  __syncthreads();
  __threadfence();                         // acquire everyone else's writes
}

__device__ inline float block_sum(float v, float* sbuf) {
  sbuf[threadIdx.x] = v;
  __syncthreads();
  for (int s = NTHR / 2; s > 0; s >>= 1) {
    if (threadIdx.x < s) sbuf[threadIdx.x] += sbuf[threadIdx.x + s];
    __syncthreads();
  }
  float r = sbuf[0];
  __syncthreads();
  return r;
}

__device__ inline float sigm(float x) { return 1.f / (1.f + __expf(-x)); }
__device__ inline float ln_apply(float v, float m, float r, float g, float b) {
  return (v - m) * r * g + b;
}

// ================= persistent LN-LSTM layer kernel =================
__global__ void __launch_bounds__(NTHR)
lstm_layer(const __bf16* __restrict__ xin_bf, const float* __restrict__ xin_f32,
           int xstride, int ktx,
           const __bf16* __restrict__ wi_t, const __bf16* __restrict__ wh_t,
           const float* __restrict__ bias,
           const float* __restrict__ gih, const float* __restrict__ bih,
           const float* __restrict__ ghh, const float* __restrict__ bhh,
           const float* __restrict__ gcn, const float* __restrict__ bcn,
           const float* __restrict__ c0, __bf16* __restrict__ hbf,
           float* __restrict__ hh_raw, float* __restrict__ xg_raw,
           float* __restrict__ out, float* __restrict__ hT, float* __restrict__ cT,
           unsigned* bar) {
  __shared__ float sred[NTHR];
  const int wg   = blockIdx.x;
  const int tid  = threadIdx.x;
  const int lane = tid & 31;
  const int wave = tid >> 5;
  const int m    = wg >> 4;             // GEMM row block (16 batch rows)
  const int n    = wg & 15;             // GEMM 256-col block
  const int nt0  = n * 16 + wave * 2;   // this wave's first 16-col tile
  const int b    = wg;                  // phase-B batch row

  float c_reg[4];
#pragma unroll
  for (int q = 0; q < 4; ++q) c_reg[q] = c0[(size_t)b * HID + q * 256 + tid];

  unsigned phase = 0;
  for (int t = 0; t < S_LEN; ++t) {
    // ---------- Phase A: recurrent GEMM + per-step input GEMM ----------
    gemm2(wh_t, hh_raw, hbf + (size_t)m * 16 * HID, nullptr, HID, HID / 32, m, nt0, lane);
    if (xin_bf)
      gemm2(wi_t, xg_raw, xin_bf + (size_t)(t * BATCH + m * 16) * xstride, nullptr,
            xstride, ktx, m, nt0, lane);
    else
      gemm2(wi_t, xg_raw, nullptr, xin_f32 + (size_t)(t * BATCH + m * 16) * xstride,
            xstride, ktx, m, nt0, lane);
    grid_sync(bar, (++phase) * NWG);

    // ---------- Phase B: LN over 4096 gates + pointwise + c-LN ----------
    float lh[16], lx[16];
    float s1h = 0, s2h = 0, s1x = 0, s2x = 0;
#pragma unroll
    for (int i = 0; i < 16; ++i) {
      float vh = hh_raw[(size_t)b * G4 + i * 256 + tid];
      float vx = xg_raw[(size_t)b * G4 + i * 256 + tid];
      lh[i] = vh; lx[i] = vx;
      s1h += vh; s2h += vh * vh; s1x += vx; s2x += vx * vx;
    }
    float mh = block_sum(s1h, sred) * (1.f / G4);
    float vh = block_sum(s2h, sred) * (1.f / G4) - mh * mh;
    float rh = rsqrtf(vh + EPS_LN);
    float mx = block_sum(s1x, sred) * (1.f / G4);
    float vx = block_sum(s2x, sred) * (1.f / G4) - mx * mx;
    float rx = rsqrtf(vx + EPS_LN);

    float o_reg[4];
    float s1c = 0, s2c = 0;
#pragma unroll
    for (int q = 0; q < 4; ++q) {
      int j0 = q * 256 + tid;                        // H index of this element
      int ji = j0, jf = j0 + 1024, jg = j0 + 2048, jo = j0 + 3072;
      float gv_i = ln_apply(lx[q],      mx, rx, gih[ji], bih[ji])
                 + ln_apply(lh[q],      mh, rh, ghh[ji], bhh[ji]) + bias[ji];
      float gv_f = ln_apply(lx[4 + q],  mx, rx, gih[jf], bih[jf])
                 + ln_apply(lh[4 + q],  mh, rh, ghh[jf], bhh[jf]) + bias[jf];
      float gv_g = ln_apply(lx[8 + q],  mx, rx, gih[jg], bih[jg])
                 + ln_apply(lh[8 + q],  mh, rh, ghh[jg], bhh[jg]) + bias[jg];
      float gv_o = ln_apply(lx[12 + q], mx, rx, gih[jo], bih[jo])
                 + ln_apply(lh[12 + q], mh, rh, ghh[jo], bhh[jo]) + bias[jo];
      float c2 = sigm(gv_f) * c_reg[q] + sigm(gv_i) * tanhf(gv_g);
      c_reg[q] = c2;
      o_reg[q] = gv_o;
      s1c += c2; s2c += c2 * c2;
    }
    float mc = block_sum(s1c, sred) * (1.f / HID);
    float vc = block_sum(s2c, sred) * (1.f / HID) - mc * mc;
    float rc = rsqrtf(vc + EPS_LN);
#pragma unroll
    for (int q = 0; q < 4; ++q) {
      int jh = q * 256 + tid;
      float cn = (c_reg[q] - mc) * rc * gcn[jh] + bcn[jh];
      float h2 = sigm(o_reg[q]) * tanhf(cn);
      out[(size_t)t * BATCH * HID + (size_t)b * HID + jh] = h2;
      hbf[(size_t)b * HID + jh] = (__bf16)h2;
      if (t == S_LEN - 1) {
        hT[(size_t)b * HID + jh] = h2;
        cT[(size_t)b * HID + jh] = c_reg[q];
      }
    }
    grid_sync(bar, (++phase) * NWG);
  }
}

// ================= prep kernels =================
// Re-tile W [G4 x Din] (row major) into per-lane-contiguous WMMA B fragments
// for B[k][n] = W[n][k]; K zero-padded to KT*32.
__global__ void tile_weights(const float* __restrict__ W, __bf16* __restrict__ outw,
                             int Din, int KT) {
  int idx = blockIdx.x * 256 + threadIdx.x;
  int total = KT * NTILES * 512;
  if (idx >= total) return;
  int e    = idx & 15;
  int lane = (idx >> 4) & 31;
  int tile = idx >> 9;
  int nt = tile % NTILES;
  int kt = tile / NTILES;
  int K = kt * 32 + (e & 7) + ((e >> 3) << 4) + ((lane >> 4) << 3);
  int N = nt * 16 + (lane & 15);
  float v = (K < Din) ? W[(size_t)N * Din + K] : 0.f;
  outw[idx] = (__bf16)v;
}

__global__ void pad_x(const float* __restrict__ x, __bf16* __restrict__ xb) {
  int idx = blockIdx.x * 256 + threadIdx.x;
  int total = S_LEN * BATCH * DIN0P;
  if (idx >= total) return;
  int d = idx % DIN0P;
  int sb = idx / DIN0P;
  xb[idx] = (d < DIN0) ? (__bf16)x[(size_t)sb * DIN0 + d] : (__bf16)0.f;
}

__global__ void conv_bf16(const float* __restrict__ src, __bf16* __restrict__ dst, int n) {
  int idx = blockIdx.x * 256 + threadIdx.x;
  if (idx < n) dst[idx] = (__bf16)src[idx];
}

// ================= host launcher =================
extern "C" void kernel_launch(void* const* d_in, const int* in_sizes, int n_in,
                              void* d_out, int out_size, void* d_ws, size_t ws_size,
                              hipStream_t stream) {
  (void)in_sizes; (void)n_in; (void)out_size; (void)ws_size;
  const float* x    = (const float*)d_in[0];
  const float* h0   = (const float*)d_in[1];
  const float* c0   = (const float*)d_in[2];
  const float* Wih0 = (const float*)d_in[3];
  const float* Wih1 = (const float*)d_in[4];
  const float* Whh  = (const float*)d_in[5];
  const float* bias = (const float*)d_in[6];
  const float* gih  = (const float*)d_in[7];
  const float* bih  = (const float*)d_in[8];
  const float* ghh  = (const float*)d_in[9];
  const float* bhh  = (const float*)d_in[10];
  const float* gc   = (const float*)d_in[11];
  const float* bc   = (const float*)d_in[12];
  float* out = (float*)d_out;

  // ---- workspace layout (~40 MB) ----
  char* ws = (char*)d_ws;
  size_t off = 0;
  auto alloc = [&](size_t bytes) -> char* {
    char* p = ws + off;
    off = (off + bytes + 255) & ~(size_t)255;
    return p;
  };
  unsigned* bar  = (unsigned*)alloc(256);
  __bf16* wi0_t  = (__bf16*)alloc((size_t)5  * NTILES * 512 * 2);  // 1.25 MB
  __bf16* wi1_t  = (__bf16*)alloc((size_t)32 * NTILES * 512 * 2);  // 8 MB
  __bf16* wh0_t  = (__bf16*)alloc((size_t)32 * NTILES * 512 * 2);  // 8 MB
  __bf16* wh1_t  = (__bf16*)alloc((size_t)32 * NTILES * 512 * 2);  // 8 MB
  __bf16* xbf    = (__bf16*)alloc((size_t)S_LEN * BATCH * DIN0P * 2); // 10 MB
  __bf16* hbf0   = (__bf16*)alloc((size_t)BATCH * HID * 2);
  __bf16* hbf1   = (__bf16*)alloc((size_t)BATCH * HID * 2);
  float*  hh_raw = (float*)alloc((size_t)BATCH * G4 * 4);           // 1 MB
  float*  xg_raw = (float*)alloc((size_t)BATCH * G4 * 4);           // 1 MB

  hipMemsetAsync(bar, 0, 256, stream);

  { int tot = 5 * NTILES * 512;
    tile_weights<<<(tot + 255) / 256, 256, 0, stream>>>(Wih0, wi0_t, DIN0, 5); }
  { int tot = 32 * NTILES * 512;
    tile_weights<<<(tot + 255) / 256, 256, 0, stream>>>(Wih1, wi1_t, HID, 32);
    tile_weights<<<(tot + 255) / 256, 256, 0, stream>>>(Whh, wh0_t, HID, 32);
    tile_weights<<<(tot + 255) / 256, 256, 0, stream>>>(Whh + (size_t)G4 * HID, wh1_t, HID, 32); }
  { int tot = S_LEN * BATCH * DIN0P;
    pad_x<<<(tot + 255) / 256, 256, 0, stream>>>(x, xbf); }
  { int tot = BATCH * HID;
    conv_bf16<<<(tot + 255) / 256, 256, 0, stream>>>(h0, hbf0, tot);
    conv_bf16<<<(tot + 255) / 256, 256, 0, stream>>>(h0 + tot, hbf1, tot); }

  float* hT = out + (size_t)S_LEN * BATCH * HID;
  float* cT = hT + 2 * BATCH * HID;

  // ---- layer 0: bf16 padded input; writes its output sequence into d_out ----
  lstm_layer<<<NWG, NTHR, 0, stream>>>(
      xbf, nullptr, DIN0P, 5,
      wi0_t, wh0_t,
      bias, gih, bih, ghh, bhh, gc, bc,
      c0, hbf0, hh_raw, xg_raw,
      out, hT, cT, bar);

  hipMemsetAsync(bar, 0, 256, stream);

  // ---- layer 1: consumes layer-0 output (fp32, in place in d_out) per step,
  //      overwriting each timestep slot with its own output after the barrier ----
  lstm_layer<<<NWG, NTHR, 0, stream>>>(
      nullptr, out, HID, 32,
      wi1_t, wh1_t,
      bias + G4, gih + G4, bih + G4, ghh + G4, bhh + G4, gc + HID, bc + HID,
      c0 + BATCH * HID, hbf1, hh_raw, xg_raw,
      out, hT + BATCH * HID, cT + BATCH * HID, bar);
}
